// FrameTransformerDecoder_63934883168455
// MI455X (gfx1250) — matrix operand order repaired and checked by reading, not compiled, not run on hardware
//
#include <hip/hip_runtime.h>

// ---------------------------------------------------------------------------
// Types for CDNA5 WMMA (wave32): bf16 A/B fragments, f32 accumulator
// ---------------------------------------------------------------------------
typedef __attribute__((ext_vector_type(16))) __bf16 v16bf;
typedef __attribute__((ext_vector_type(8)))  __bf16 v8bf;
typedef __attribute__((ext_vector_type(8)))  float  v8f;

#define GTM 128
#define GTN 128
#define GTK 32
#define LSTR 40   // LDS row stride in bf16 elems (80B) -> 16B aligned, conflict-padded

// ---------------------------------------------------------------------------
// bf16 GEMM:  C[M,N] = alpha*(A[M,K] x B(^T) + addA) + addB   (fp32 out)
//   TRANSB template: 1 -> B is [N,K], 0 -> B is [K,N]
//   batched over gridDim.z with element strides sA/sB/sC (sAA/sAB for adds)
//   assumes M%128==0, N%128==0, K%32==0 (true for every call below)
//   Double-buffered LDS: global prefetch of tile k+1 overlaps 8 WMMAs of tile k.
// ---------------------------------------------------------------------------
template <int TRANSB>
__global__ __launch_bounds__(256) void k_gemm_t(
    const __bf16* __restrict__ A, long long sA, int lda,
    const __bf16* __restrict__ B, long long sB, int ldb,
    float* __restrict__ C, long long sC, int ldc,
    const float* __restrict__ addA, long long sAA,
    const float* __restrict__ addB, long long sAB,
    float alpha, int relu, int M, int N, int K)
{
  __shared__ __bf16 As[2][GTM * LSTR];
  __shared__ __bf16 Bs[2][GTN * LSTR];

  const int z = blockIdx.z;
  const __bf16* Ab = A + (long long)z * sA;
  const __bf16* Bb = B + (long long)z * sB;
  float* Cb = C + (long long)z * sC;

  const int tid   = threadIdx.x;
  const int lane  = tid & 31;
  const int wave  = tid >> 5;
  const int waveM = wave >> 2;     // 0..1 : 64 rows
  const int waveN = wave & 3;      // 0..3 : 32 cols
  const int l16   = lane & 15;
  const int half  = lane >> 4;

  const int tileM = blockIdx.y * GTM;
  const int tileN = blockIdx.x * GTN;

  v8f acc[4][2];
#pragma unroll
  for (int mi = 0; mi < 4; ++mi)
#pragma unroll
    for (int ni = 0; ni < 2; ++ni)
#pragma unroll
      for (int e = 0; e < 8; ++e) acc[mi][ni][e] = 0.f;

  const int aRow = tid >> 1;
  const int aCol = (tid & 1) << 4;
  const int bK   = tid >> 3;         // for TRANSB==0
  const int bN   = (tid & 7) << 4;

  v8bf a0, a1, b0, b1;

  auto stageLoad = [&](int k0) {
    const __bf16* srcA = Ab + (long long)(tileM + aRow) * lda + (k0 + aCol);
    a0 = *(const v8bf*)(srcA);
    a1 = *(const v8bf*)(srcA + 8);
    if (TRANSB) {
      const __bf16* srcB = Bb + (long long)(tileN + aRow) * ldb + (k0 + aCol);
      b0 = *(const v8bf*)(srcB);
      b1 = *(const v8bf*)(srcB + 8);
    } else {
      const __bf16* srcB = Bb + (long long)(k0 + bK) * ldb + (tileN + bN);
      b0 = *(const v8bf*)(srcB);
      b1 = *(const v8bf*)(srcB + 8);
    }
  };
  auto stageStore = [&](int buf) {
    *(v8bf*)(&As[buf][aRow * LSTR + aCol])     = a0;
    *(v8bf*)(&As[buf][aRow * LSTR + aCol + 8]) = a1;
    if (TRANSB) {
      *(v8bf*)(&Bs[buf][aRow * LSTR + aCol])     = b0;
      *(v8bf*)(&Bs[buf][aRow * LSTR + aCol + 8]) = b1;
    } else {
#pragma unroll
      for (int e = 0; e < 8; ++e) {
        Bs[buf][(bN + e) * LSTR + bK]     = b0[e];
        Bs[buf][(bN + 8 + e) * LSTR + bK] = b1[e];
      }
    }
  };

  stageLoad(0);
  stageStore(0);
  __syncthreads();

  int buf = 0;
  for (int k0 = 0; k0 < K; k0 += GTK) {
    const bool hasNext = (k0 + GTK) < K;
    if (hasNext) stageLoad(k0 + GTK);   // global -> regs, overlaps WMMA below

    // A fragment (16x32 bf16): lane l16=row, halves hold K 0..7/16..23 | 8..15/24..31
    v16bf af[4], bfg[2];
#pragma unroll
    for (int mi = 0; mi < 4; ++mi) {
      const __bf16* p = &As[buf][(waveM * 64 + mi * 16 + l16) * LSTR + half * 8];
      v8bf lo = *(const v8bf*)p;
      v8bf hi = *(const v8bf*)(p + 16);
#pragma unroll
      for (int e = 0; e < 8; ++e) { af[mi][e] = lo[e]; af[mi][e + 8] = hi[e]; }
    }
    // B fragment: lane l16=col, halves hold K 0..7/16..23 | 8..15/24..31
#pragma unroll
    for (int ni = 0; ni < 2; ++ni) {
      const __bf16* p = &Bs[buf][(waveN * 32 + ni * 16 + l16) * LSTR + half * 8];
      v8bf lo = *(const v8bf*)p;
      v8bf hi = *(const v8bf*)(p + 16);
#pragma unroll
      for (int e = 0; e < 8; ++e) { bfg[ni][e] = lo[e]; bfg[ni][e + 8] = hi[e]; }
    }

#pragma unroll
    for (int mi = 0; mi < 4; ++mi)
#pragma unroll
      for (int ni = 0; ni < 2; ++ni)
        acc[mi][ni] = __builtin_amdgcn_wmma_f32_16x16x32_bf16(
            false, af[mi], false, bfg[ni], (short)0, acc[mi][ni], false, false);

    if (hasNext) stageStore(buf ^ 1);
    __syncthreads();
    buf ^= 1;
  }

  // epilogue; C/D layout: lane covers col=l16, VGPR r -> row = r + 8*half
  const bool hasAA = addA != nullptr;
  const bool hasAB = addB != nullptr;
#pragma unroll
  for (int mi = 0; mi < 4; ++mi) {
#pragma unroll
    for (int ni = 0; ni < 2; ++ni) {
      const int col = tileN + waveN * 32 + ni * 16 + l16;
#pragma unroll
      for (int r = 0; r < 8; ++r) {
        const int row = tileM + waveM * 64 + mi * 16 + half * 8 + r;
        const long long idx = (long long)row * ldc + col;
        float v = acc[mi][ni][r];
        if (hasAA) v += addA[(long long)z * sAA + idx];
        v *= alpha;
        if (hasAB) v += addB[(long long)z * sAB + idx];
        if (relu) v = fmaxf(v, 0.f);
        Cb[idx] = v;
      }
    }
  }
}

// ---------------------------------------------------------------------------
// Elementwise / helper kernels
// ---------------------------------------------------------------------------
__global__ void k_f32_to_bf16(const float* __restrict__ s, __bf16* __restrict__ d, long long n) {
  long long i = (long long)blockIdx.x * 256 + threadIdx.x;
  if (i < n) d[i] = (__bf16)s[i];
}

// relw [4][256][1024] f32 -> relwT [4][1024][256] bf16
__global__ void k_relw_t(const float* __restrict__ relw, __bf16* __restrict__ out, int total) {
  int i = blockIdx.x * 256 + threadIdx.x;
  if (i >= total) return;
  int d = i & 255, t = (i >> 8) & 1023, h = i >> 18;
  out[i] = (__bf16)relw[((h << 8) + d) * 1024 + t];
}

// xh[b][w][h] = sum_c x[b][c][h][w] * Win[c]   (LDS-tiled transpose)
__global__ __launch_bounds__(256) void k_inproj(const float* __restrict__ x,
                                                const float* __restrict__ Win,
                                                float* __restrict__ xh) {
  __shared__ float t[32][33];
  const int b = blockIdx.z, h0 = blockIdx.y * 32, w0 = blockIdx.x * 32;
  const int col = threadIdx.x & 31, rw = threadIdx.x >> 5;
  float wc[8];
#pragma unroll
  for (int c = 0; c < 8; ++c) wc[c] = Win[c];
#pragma unroll
  for (int l = 0; l < 4; ++l) {
    const int hh = rw + l * 8;
    float s = 0.f;
#pragma unroll
    for (int c = 0; c < 8; ++c)
      s += x[(((long long)b * 8 + c) * 1024 + (h0 + hh)) * 1024 + (w0 + col)] * wc[c];
    t[hh][col] = s;
  }
  __syncthreads();
#pragma unroll
  for (int l = 0; l < 4; ++l) {
    const int ww = rw + l * 8;
    xh[((long long)b * 1024 + (w0 + ww)) * 1024 + (h0 + col)] = t[col][ww];
  }
}

// out[b][h][w] = in[b][w][h]
__global__ __launch_bounds__(256) void k_transpose(const float* __restrict__ in, float* __restrict__ out) {
  __shared__ float t[32][33];
  const int b = blockIdx.z, w0 = blockIdx.x * 32, h0 = blockIdx.y * 32;
  const int col = threadIdx.x & 31, rw = threadIdx.x >> 5;
#pragma unroll
  for (int l = 0; l < 4; ++l) {
    const int ww = rw + l * 8;
    t[ww][col] = in[((long long)b * 1024 + (w0 + ww)) * 1024 + (h0 + col)];
  }
  __syncthreads();
#pragma unroll
  for (int l = 0; l < 4; ++l) {
    const int hh = rw + l * 8;
    out[((long long)b * 1024 + (h0 + hh)) * 1024 + (w0 + col)] = t[col][hh];
  }
}

// LayerNorm over rows of 1024; optional fp32 and/or bf16 outputs
__global__ __launch_bounds__(256) void k_layernorm(const float* __restrict__ x,
                                                   const float* __restrict__ g,
                                                   const float* __restrict__ b,
                                                   float* __restrict__ outF,
                                                   __bf16* __restrict__ outH) {
  const long long row = blockIdx.x;
  const float* xr = x + row * 1024;
  float v[4], s = 0.f, s2 = 0.f;
#pragma unroll
  for (int k = 0; k < 4; ++k) { v[k] = xr[threadIdx.x + k * 256]; s += v[k]; s2 += v[k] * v[k]; }
  __shared__ float r1[256], r2[256];
  r1[threadIdx.x] = s; r2[threadIdx.x] = s2;
  __syncthreads();
  for (int o = 128; o > 0; o >>= 1) {
    if (threadIdx.x < o) { r1[threadIdx.x] += r1[threadIdx.x + o]; r2[threadIdx.x] += r2[threadIdx.x + o]; }
    __syncthreads();
  }
  const float mean = r1[0] * (1.f / 1024.f);
  const float var  = r2[0] * (1.f / 1024.f) - mean * mean;
  const float rs   = rsqrtf(var + 1e-5f);
#pragma unroll
  for (int k = 0; k < 4; ++k) {
    const int c = threadIdx.x + k * 256;
    const float o = (v[k] - mean) * rs * g[c] + b[c];
    if (outF) outF[row * 1024 + c] = o;
    if (outH) outH[row * 1024 + c] = (__bf16)o;
  }
}

// depthwise conv over time; in [b,w,C] f32; optional head-split bf16 output
__global__ __launch_bounds__(256) void k_dwconv(const float* __restrict__ in,
                                                const float* __restrict__ wt,
                                                int ksize, int pad, int Wlen, int C, long long total,
                                                float* __restrict__ outF, __bf16* __restrict__ outH,
                                                int split) {
  long long i = (long long)blockIdx.x * 256 + threadIdx.x;
  if (i >= total) return;
  int c = (int)(i % C);
  int w = (int)((i / C) % Wlen);
  long long b = i / ((long long)C * Wlen);
  float s = 0.f;
  for (int j = 0; j < ksize; ++j) {
    int ww = w + j - pad;
    if (ww >= 0 && ww < Wlen) s += in[(b * Wlen + ww) * C + c] * wt[c * ksize + j];
  }
  if (outF) outF[i] = s;
  if (outH) {
    long long oi = i;
    if (split) {
      int D = C >> 2, n = c / D, d = c - n * D;
      oi = ((b * 4 + n) * (long long)Wlen + w) * D + d;
    }
    outH[oi] = (__bf16)s;
  }
}

// row softmax over 1024; fp32 in -> bf16 probs
__global__ __launch_bounds__(256) void k_softmax(const float* __restrict__ qk, __bf16* __restrict__ p) {
  const long long row = blockIdx.x;
  const float* xr = qk + row * 1024;
  __shared__ float red[256];
  float vals[4], m = -1e30f;
#pragma unroll
  for (int k = 0; k < 4; ++k) { vals[k] = xr[threadIdx.x + k * 256]; m = fmaxf(m, vals[k]); }
  red[threadIdx.x] = m; __syncthreads();
  for (int o = 128; o > 0; o >>= 1) {
    if (threadIdx.x < o) red[threadIdx.x] = fmaxf(red[threadIdx.x], red[threadIdx.x + o]);
    __syncthreads();
  }
  m = red[0]; __syncthreads();
  float s = 0.f;
#pragma unroll
  for (int k = 0; k < 4; ++k) { vals[k] = __expf(vals[k] - m); s += vals[k]; }
  red[threadIdx.x] = s; __syncthreads();
  for (int o = 128; o > 0; o >>= 1) {
    if (threadIdx.x < o) red[threadIdx.x] += red[threadIdx.x + o];
    __syncthreads();
  }
  const float inv = 1.f / red[0];
#pragma unroll
  for (int k = 0; k < 4; ++k)
    p[row * 1024 + threadIdx.x + k * 256] = (__bf16)(vals[k] * inv);
}

// av f32 [b*4+n][w][256] -> am bf16 [b][w][n*256+d]
__global__ void k_merge(const float* __restrict__ av, __bf16* __restrict__ am, long long total) {
  long long i = (long long)blockIdx.x * 256 + threadIdx.x;
  if (i >= total) return;
  int c = (int)(i & 1023);
  long long w = (i >> 10) & 1023, b = i >> 20;
  int n = c >> 8, d = c & 255;
  am[i] = (__bf16)av[(((b * 4 + n) * 1024) + w) * 256 + d];
}

// hL[..., 0:256] += hR
__global__ void k_addslice(float* __restrict__ hL, const float* __restrict__ hR, long long total) {
  long long i = (long long)blockIdx.x * 256 + threadIdx.x;
  if (i >= total) return;
  long long r = i >> 8; int c = (int)(i & 255);
  hL[r * 1024 + c] += hR[i];
}

// t2 = bf16( relu(t)^2 )
__global__ void k_relusq(const float* __restrict__ t, __bf16* __restrict__ o, long long n) {
  long long i = (long long)blockIdx.x * 256 + threadIdx.x;
  if (i >= n) return;
  float v = fmaxf(t[i], 0.f);
  o[i] = (__bf16)(v * v);
}

// ---------------------------------------------------------------------------
// Host orchestration
// ---------------------------------------------------------------------------
extern "C" void kernel_launch(void* const* d_in, const int* in_sizes, int n_in,
                              void* d_out, int out_size, void* d_ws, size_t ws_size,
                              hipStream_t stream) {
  (void)in_sizes; (void)n_in; (void)out_size; (void)ws_size;

  const float* x        = (const float*)d_in[0];
  const float* prev_qk1 = (const float*)d_in[2];
  const float* prev_qk2 = (const float*)d_in[3];
  const float* Win      = (const float*)d_in[4];
  const float* ln_g     = (const float*)d_in[6];
  const float* ln_b     = (const float*)d_in[7];
  const float* ln3_g    = (const float*)d_in[8];
  const float* ln3_b    = (const float*)d_in[9];
  const float* aWq      = (const float*)d_in[10];
  const float* aWk      = (const float*)d_in[11];
  const float* aWv      = (const float*)d_in[12];
  const float* aWo      = (const float*)d_in[13];
  const float* aqc      = (const float*)d_in[14];
  const float* akc      = (const float*)d_in[15];
  const float* avc      = (const float*)d_in[16];
  const float* arelw    = (const float*)d_in[17];
  const float* dw11     = (const float*)d_in[18];
  const float* pw1L     = (const float*)d_in[19];
  const float* dw7a     = (const float*)d_in[20];
  const float* pw1R     = (const float*)d_in[21];
  const float* dw7b     = (const float*)d_in[22];
  const float* pw2      = (const float*)d_in[23];
  const float* W3       = (const float*)d_in[24];
  const float* W4       = (const float*)d_in[25];

  char* ws = (char*)d_ws;
  size_t off = 0;
  auto A8 = [&](size_t bytes) {
    size_t r = (off + 255) & ~(size_t)255;
    off = r + bytes;
    return (void*)(ws + r);
  };

  __bf16* WQb   = (__bf16*)A8(8388608);
  __bf16* WKb   = (__bf16*)A8(8388608);
  __bf16* WVb   = (__bf16*)A8(8388608);
  __bf16* WOb   = (__bf16*)A8(8388608);
  __bf16* PW1Lb = (__bf16*)A8(2097152);
  __bf16* PW1Rb = (__bf16*)A8(524288);
  __bf16* PW2b  = (__bf16*)A8(2097152);
  __bf16* W3b   = (__bf16*)A8(4194304);
  __bf16* W4b   = (__bf16*)A8(4194304);
  __bf16* RELWT = (__bf16*)A8(2097152);
  float*  XH    = (float*) A8(8388608);   // [2,1024,1024]
  float*  HLNF  = (float*) A8(8388608);
  __bf16* HLNH  = (__bf16*)A8(4194304);
  float*  UF    = (float*) A8(8388608);   // proj intermediates / FFN hL
  __bf16* QH    = (__bf16*)A8(4194304);   // [8,1024,256]
  __bf16* KH    = (__bf16*)A8(4194304);
  __bf16* VH    = (__bf16*)A8(4194304);
  float*  PT    = (float*) A8(33554432);  // rel bias [8,1024,1024]
  float*  S1    = (float*) A8(33554432);  // qk of attn0 (prev for attn2)
  float*  S2    = (float*) A8(33554432);  // qk of attn1 (prev for attn3)
  __bf16* PROBS = (__bf16*)A8(16777216);
  float*  AV    = (float*) A8(8388608);
  __bf16* AM    = (__bf16*)A8(4194304);
  __bf16* D1    = (__bf16*)A8(4194304);
  float*  HR    = (float*) A8(2097152);
  float*  TF    = PT;                 // reuse: final MLP hidden fp32 [2,1024,2048]
  __bf16* T2    = PROBS;              // reuse: bf16 hidden

  float* out_xh   = (float*)d_out;
  float* out_pqk1 = out_xh + 2097152;
  float* out_pqk2 = out_xh + 10485760;

  auto cvt = [&](const float* s, __bf16* d, long long n) {
    k_f32_to_bf16<<<(unsigned)((n + 255) / 256), 256, 0, stream>>>(s, d, n);
  };
  auto gemm = [&](const __bf16* A, long long sA, int lda,
                  const __bf16* B, long long sB, int ldb, int transB,
                  float* C, long long sC, int ldc,
                  const float* adA, long long sAA,
                  const float* adB, long long sAB,
                  float alpha, int relu, int M, int N, int K, int batch) {
    dim3 g(N / 128, M / 128, batch);
    if (transB)
      k_gemm_t<1><<<g, 256, 0, stream>>>(A, sA, lda, B, sB, ldb,
                                         C, sC, ldc, adA, sAA, adB, sAB,
                                         alpha, relu, M, N, K);
    else
      k_gemm_t<0><<<g, 256, 0, stream>>>(A, sA, lda, B, sB, ldb,
                                         C, sC, ldc, adA, sAA, adB, sAB,
                                         alpha, relu, M, N, K);
  };
  auto ln = [&](const float* src, const float* g, const float* b, float* oF, __bf16* oH) {
    k_layernorm<<<2048, 256, 0, stream>>>(src, g, b, oF, oH);
  };

  // ---- weight preparation (bf16) ----
  cvt(aWq, WQb, 4194304); cvt(aWk, WKb, 4194304);
  cvt(aWv, WVb, 4194304); cvt(aWo, WOb, 4194304);
  cvt(pw1L, PW1Lb, 1048576); cvt(pw1R, PW1Rb, 262144); cvt(pw2, PW2b, 1048576);
  cvt(W3, W3b, 2097152);  cvt(W4, W4b, 2097152);
  k_relw_t<<<4096, 256, 0, stream>>>(arelw, RELWT, 1048576);

  // ---- input projection: xh[b,w,h] ----
  k_inproj<<<dim3(32, 32, 2), 256, 0, stream>>>(x, Win, XH);

  // ---- attention block ----
  auto attn = [&](int i, const float* prev, float* qkdst) {
    const __bf16* Wp[3]  = {WQb + (size_t)i * 1048576, WKb + (size_t)i * 1048576, WVb + (size_t)i * 1048576};
    const float*  cp[3]  = {aqc + i * 3072, akc + i * 3072, avc + i * 3072};
    __bf16*       op[3]  = {QH, KH, VH};
    for (int p = 0; p < 3; ++p) {
      gemm(HLNH, 0, 1024, Wp[p], 0, 1024, 1, UF, 0, 1024,
           nullptr, 0, nullptr, 0, 1.f, 0, 2048, 1024, 1024, 1);
      k_dwconv<<<8192, 256, 0, stream>>>(UF, cp[p], 3, 1, 1024, 1024, 2097152LL,
                                         nullptr, op[p], 1);
    }
    // rel bias transposed: pT[t][j] = sum_d relwT[t][d]*q[j][d]
    gemm(RELWT + (size_t)i * 262144, 0, 256, QH, 262144, 256, 1,
         PT, 1048576, 1024, nullptr, 0, nullptr, 0, 1.f, 0, 1024, 1024, 256, 8);
    // qk = (q k^T + pT)/sqrt(1024) + prev
    gemm(QH, 262144, 256, KH, 262144, 256, 1,
         qkdst, 1048576, 1024, PT, 1048576, prev, 1048576,
         0.03125f, 0, 1024, 1024, 256, 8);
    k_softmax<<<8192, 256, 0, stream>>>(qkdst, PROBS);
    // av = softmax(qk) @ v   (B non-transposed [K=w, N=d])
    gemm(PROBS, 1048576, 1024, VH, 262144, 256, 0,
         AV, 262144, 256, nullptr, 0, nullptr, 0, 1.f, 0, 1024, 256, 1024, 8);
    k_merge<<<8192, 256, 0, stream>>>(AV, AM, 2097152LL);
    // xh += am @ Wo^T (in-place residual)
    gemm(AM, 0, 1024, WOb + (size_t)i * 1048576, 0, 1024, 1,
         XH, 0, 1024, nullptr, 0, XH, 0, 1.f, 0, 2048, 1024, 1024, 1);
  };

  // block 1: two parallel attentions on same LN(xh)
  ln(XH, ln_g, ln_b, nullptr, HLNH);
  attn(0, prev_qk1, S1);
  attn(1, prev_qk2, S2);

  // conv-FFN
  ln(XH, ln_g + 1024, ln_b + 1024, HLNF, nullptr);
  k_dwconv<<<8192, 256, 0, stream>>>(HLNF, dw11, 11, 5, 1024, 1024, 2097152LL, nullptr, D1, 0);
  gemm(D1, 0, 1024, PW1Lb, 0, 1024, 1, UF, 0, 1024,
       nullptr, 0, nullptr, 0, 1.f, 1, 2048, 1024, 1024, 1);           // hL = relu(...)
  k_dwconv<<<8192, 256, 0, stream>>>(HLNF, dw7a, 7, 3, 1024, 1024, 2097152LL, nullptr, D1, 0);
  gemm(D1, 0, 1024, PW1Rb, 0, 1024, 1, HR, 0, 256,
       nullptr, 0, nullptr, 0, 1.f, 0, 2048, 256, 1024, 1);            // hR
  k_addslice<<<2048, 256, 0, stream>>>(UF, HR, 524288LL);
  ln(UF, ln3_g, ln3_b, HLNF, nullptr);                                 // LN3
  k_dwconv<<<8192, 256, 0, stream>>>(HLNF, dw7b, 7, 3, 1024, 1024, 2097152LL, nullptr, D1, 0);
  gemm(D1, 0, 1024, PW2b, 0, 1024, 1, XH, 0, 1024,
       nullptr, 0, XH, 0, 1.f, 0, 2048, 1024, 1024, 1);                // xh += ...

  // attentions 2 & 3 (qk written straight to d_out pqk regions)
  ln(XH, ln_g + 2048, ln_b + 2048, nullptr, HLNH);
  attn(2, S1, out_pqk1);
  ln(XH, ln_g + 3072, ln_b + 3072, nullptr, HLNH);
  attn(3, S2, out_pqk2);

  // final squared-ReLU MLP
  ln(XH, ln_g + 4096, ln_b + 4096, nullptr, HLNH);
  gemm(HLNH, 0, 1024, W3b, 0, 1024, 1, TF, 0, 2048,
       nullptr, 0, nullptr, 0, 1.f, 0, 2048, 2048, 1024, 1);
  k_relusq<<<16384, 256, 0, stream>>>(TF, T2, 4194304LL);
  gemm(T2, 0, 2048, W4b, 0, 2048, 1, XH, 0, 1024,
       nullptr, 0, XH, 0, 1.f, 0, 2048, 1024, 2048, 1);

  // output: xh.transpose(0,2,1)
  k_transpose<<<dim3(32, 32, 2), 256, 0, stream>>>(XH, out_xh);
}
